// SegmentedAttention_52166672777374
// MI455X (gfx1250) — matrix operand order repaired and checked by reading, not compiled, not run on hardware
//
#include <hip/hip_runtime.h>

typedef __attribute__((ext_vector_type(16))) __bf16 v16bf;
typedef __attribute__((ext_vector_type(8)))  float  v8f;
typedef __attribute__((ext_vector_type(4)))  unsigned int v4u;
typedef __attribute__((ext_vector_type(8)))  int    v8i_;
typedef __attribute__((ext_vector_type(4)))  int    v4i_;
typedef unsigned short u16;
typedef unsigned int   u32;

// ---------- helpers ----------

__device__ __forceinline__ u16 f2bf(float f) {
    u32 u = __float_as_uint(f);
    u32 r = (u + 0x7FFFu + ((u >> 16) & 1u)) >> 16;   // round-to-nearest-even
    return (u16)r;
}

union Frag32B {
    uint4 q[2];
    u16   h[16];
    v16bf v;
};

// Build a 16-element bf16 fragment from two 16B chunks (A-operand pattern).
__device__ __forceinline__ v16bf load_ab(const u16* p0, const u16* p1) {
    Frag32B f;
    f.q[0] = *reinterpret_cast<const uint4*>(p0);
    f.q[1] = *reinterpret_cast<const uint4*>(p1);
    return f.v;
}
// 32B contiguous chunk (B-operand pattern).
__device__ __forceinline__ v16bf load16c(const u16* p) { return load_ab(p, p + 8); }

__device__ __forceinline__ v8f wmma_bf16(v16bf a, v16bf b, v8f c) {
    return __builtin_amdgcn_wmma_f32_16x16x32_bf16(false, a, false, b, (short)0, c, false, false);
}

// ---------- Tensor Data Mover (gfx1250 TDM) ----------
#if defined(__has_builtin)
#if __has_builtin(__builtin_amdgcn_tensor_load_to_lds)
#define HAVE_TDM 1
#endif
#endif
#ifndef HAVE_TDM
#define HAVE_TDM 0
#endif

// compile-log probes (visible in stderr even on successful compile)
#if HAVE_TDM
#if __clang_major__ >= 23
#warning "CDNA5 probe: using 6-arg __builtin_amdgcn_tensor_load_to_lds (clang >= 23)"
#else
#warning "CDNA5 probe: using 5-arg __builtin_amdgcn_tensor_load_to_lds (clang < 23)"
#endif
#else
#warning "CDNA5 probe: tensor_load_to_lds builtin NOT available - cooperative LDS copy fallback in flash kernel"
#endif

__device__ __forceinline__ u32 lds_off(const void* p) {
    // generic pointer to LDS: low 32 bits are the LDS byte offset
    return (u32)(unsigned long long)p;
}

// 2D tile load: tile_w x tile_h (16-bit elements), row stride in elements.
__device__ __forceinline__ void tdm_load_2d(u32 ldsaddr, const void* gptr,
                                            u32 tile_w, u32 tile_h,
                                            u32 tensor_w, u32 tensor_h,
                                            u32 row_stride) {
#if HAVE_TDM
    unsigned long long ga = (unsigned long long)gptr;
    v4u g0;
    g0.x = 1u;                                  // count=1, user descriptor
    g0.y = ldsaddr;                             // lds_addr (bytes)
    g0.z = (u32)ga;                             // global_addr[31:0]
    g0.w = (u32)(ga >> 32) | (2u << 30);        // global_addr[56:32] | type=2
    v8i_ g1;
    g1[0] = 0x10000;                                          // data_size=1 (2B), wg_mask=0
    g1[1] = (int)((tensor_w & 0xFFFFu) << 16);                // tensor_dim0 lo<<16
    g1[2] = (int)((tensor_w >> 16) | ((tensor_h & 0xFFFFu) << 16)); // dim0 hi | dim1 lo
    g1[3] = (int)((tensor_h >> 16) | (tile_w << 16));         // dim1 hi | tile_dim0
    g1[4] = (int)tile_h;                                      // tile_dim1 (tile_dim2=0)
    g1[5] = (int)row_stride;                                  // tensor_dim0_stride[31:0]
    g1[6] = 0;
    g1[7] = 0;
    v4i_ z4 = {0, 0, 0, 0};
#if __clang_major__ >= 23
    v8i_ z8 = {0, 0, 0, 0, 0, 0, 0, 0};
    __builtin_amdgcn_tensor_load_to_lds(g0, g1, z4, z4, z8, 0);
#else
    __builtin_amdgcn_tensor_load_to_lds(g0, g1, z4, z4, 0);
#endif
#else
    (void)ldsaddr; (void)gptr; (void)tile_w; (void)tile_h;
    (void)tensor_w; (void)tensor_h; (void)row_stride;
#endif
}

// ---------- constants ----------
// B=2 S=8192 DIM=512 HEADS=8 DH=64 SEG=512 NPM=4 -> 32 segments, M=16384 rows.
// k extended rows: 4 pmem + 512 + 28 pad = 544 = 17*32.
#define KEXT_R   544
#define VOUT_OFF 8388608

// ---------- prep kernels ----------

__global__ __launch_bounds__(256) void prep_x_kernel(const float* __restrict__ seq,
                                                     const float* __restrict__ w,
                                                     u16* __restrict__ xb) {
    const int wave = threadIdx.x >> 5, lane = threadIdx.x & 31;
    const int row  = blockIdx.x * 8 + wave;
    const float* s = seq + (size_t)row * 512;
    float acc = 0.f;
    for (int e = lane; e < 512; e += 32) { float v = s[e]; acc += v * v; }
#pragma unroll
    for (int d = 16; d >= 1; d >>= 1) acc += __shfl_xor(acc, d, 32);
    const float scale = rsqrtf(acc * (1.0f / 512.0f) + 1.1920928955078125e-7f);
    for (int e = lane; e < 512; e += 32) xb[(size_t)row * 512 + e] = f2bf(s[e] * scale * w[e]);
}

// wt[n*rowsK + k] = w[k*cols + n]
__global__ void prep_wT_kernel(const float* __restrict__ w, u16* __restrict__ wt,
                               int rowsK, int cols) {
    int idx = blockIdx.x * blockDim.x + threadIdx.x;
    if (idx >= rowsK * cols) return;
    int n = idx / rowsK, k = idx % rowsK;
    wt[idx] = f2bf(w[(size_t)k * cols + n]);
}

// k_ext: row-major [sh][544][64]; vT: transposed [sh][64][544].
// fill pmem rows 0..3 and zero pad rows 516..543 of both.
__global__ void prep_pmem_kernel(const float* __restrict__ pmem,
                                 u16* __restrict__ kext, u16* __restrict__ vextT) {
    int idx = blockIdx.x * blockDim.x + threadIdx.x;
    if (idx >= 32 * 8 * 32 * 64) return;
    int d = idx & 63; int t = idx >> 6;
    int rr = t & 31; t >>= 5;
    int h  = t & 7;  int seg = t >> 3;
    int row = (rr < 4) ? rr : (512 + rr);       // 4..31 -> 516..543
    float kv = 0.f, vv = 0.f;
    if (rr < 4) {
        kv = pmem[((0 * 8 + h) * 4 + rr) * 64 + d];
        vv = pmem[((1 * 8 + h) * 4 + rr) * 64 + d];
    }
    const int sh = seg * 8 + h;
    kext[((size_t)sh * KEXT_R + row) * 64 + d] = f2bf(kv);
    vextT[((size_t)sh * 64 + d) * KEXT_R + row] = f2bf(vv);
}

// ---------- QKV GEMM: x(16384x512) @ WqkvT -> q / k(rotary) / v ----------
// block 256 = 8 waves; block tile 256(M) x 64(N); wave tile 32x64 (8 wmma accs).

__global__ __launch_bounds__(256) void qkv_gemm_kernel(const u16* __restrict__ x,
                                                       const u16* __restrict__ wqt,
                                                       u16* __restrict__ qarr,
                                                       u16* __restrict__ kext,
                                                       u16* __restrict__ vextT,
                                                       float* __restrict__ vout) {
    const int wave = threadIdx.x >> 5, lane = threadIdx.x & 31;
    const int lm = lane & 15, lh = lane >> 4;
    const int m0 = blockIdx.x * 256 + wave * 32;
    const int n0 = blockIdx.y * 64;
    const u16* arow0 = x + (size_t)(m0 + lm) * 512;
    const u16* arow1 = x + (size_t)(m0 + 16 + lm) * 512;
    v8f acc[2][4] = {};
    for (int ks = 0; ks < 16; ++ks) {
        const u16* ap0 = arow0 + ks * 32;
        const u16* ap1 = arow1 + ks * 32;
        if (ks + 1 < 16) { __builtin_prefetch(ap0 + 32, 0, 3); __builtin_prefetch(ap1 + 32, 0, 3); }
        v16bf af0 = load_ab(ap0 + lh * 16, ap0 + lh * 16 + 8);
        v16bf af1 = load_ab(ap1 + lh * 16, ap1 + lh * 16 + 8);
#pragma unroll
        for (int nb = 0; nb < 4; ++nb) {
            const u16* bp = wqt + (size_t)(n0 + nb * 16 + lm) * 512 + ks * 32 + lh * 16;
            v16bf bf = load16c(bp);
            acc[0][nb] = wmma_bf16(af0, bf, acc[0][nb]);
            acc[1][nb] = wmma_bf16(af1, bf, acc[1][nb]);
        }
    }
#pragma unroll
    for (int mi = 0; mi < 2; ++mi) {
        const int mb = m0 + mi * 16 + 8 * lh;
#pragma unroll
        for (int nb = 0; nb < 4; ++nb) {
            const int c = n0 + nb * 16 + lm;
#pragma unroll
            for (int p = 0; p < 8; ++p) {
                const int m = mb + p;
                const int seg = m >> 9;
                const int i = m & 511;
                float val = acc[mi][nb][p];
                if (c < 1024) {
                    // rotary (interleaved pairs); partner column c^1 lives in lane^1
                    const int cc = c & 63;
                    const int j = cc >> 1;
                    float inv_freq = __expf(-(float)(2 * j) * (9.210340371976184f / 64.0f));
                    float ang = (float)i * inv_freq;
                    float sa, ca; __sincosf(ang, &sa, &ca);
                    float pr = __shfl_xor(val, 1, 32);
                    float rot = (cc & 1) ? pr : -pr;
                    float rv = val * ca + rot * sa;
                    const int hd = (c >> 6) & 7;
                    if (c < 512) {
                        qarr[((size_t)(seg * 8 + hd) * 512 + i) * 64 + cc] = f2bf(rv);
                    } else {
                        kext[((size_t)(seg * 8 + hd) * KEXT_R + 4 + i) * 64 + cc] = f2bf(rv);
                    }
                } else {
                    const int cv = c - 1024;
                    const int hd = cv >> 6, cd = cv & 63;
                    vextT[((size_t)(seg * 8 + hd) * 64 + cd) * KEXT_R + 4 + i] = f2bf(val);
                    vout[((size_t)(seg * 8 + hd) * 512 + i) * 64 + cd] = val;   // orig_v fp32
                }
            }
        }
    }
}

// ---------- flash attention per (seg,head) x 64-row q block ----------
// block 128 = 4 waves. K blocks (32x64 bf16, 4KB) staged into LDS by the TDM,
// double-buffered; V^T operands load straight from global (contiguous 32B).

__global__ __launch_bounds__(128) void flash_kernel(const u16* __restrict__ qarr,
                                                    const u16* __restrict__ kext,
                                                    const u16* __restrict__ vextT,
                                                    u16* __restrict__ aout) {
    __shared__ u16 kbuf[2][32 * 64];   // K tile double buffer
    __shared__ u16 ps[4][16 * 32];     // per-wave P-tile scratch
    const int sh = blockIdx.x;         // seg*8 + head
    const int seg = sh >> 3, head = sh & 7;
    const int qbase = blockIdx.y * 64;
    const int wave = threadIdx.x >> 5, lane = threadIdx.x & 31;
    const int lm = lane & 15, lh = lane >> 4;
    const u16* qp  = qarr  + (size_t)sh * 512 * 64;
    const u16* kp  = kext  + (size_t)sh * KEXT_R * 64;  // row-major [544][64]
    const u16* vtp = vextT + (size_t)sh * 64 * KEXT_R;  // transposed [64][544]

    const int qrow = qbase + wave * 16 + lm;
    const u16* qr = qp + (size_t)qrow * 64;
    v16bf qf0 = load_ab(qr + lh * 16, qr + lh * 16 + 8);
    v16bf qf1 = load_ab(qr + 32 + lh * 16, qr + 40 + lh * 16);

    v8f o0 = {}, o1 = {}, o2 = {}, o3 = {};
    float mr[8], lr[8];
#pragma unroll
    for (int p = 0; p < 8; ++p) { mr[p] = -1e30f; lr[p] = 0.f; }

    int nkb = (qbase + 67) / 32 + 1;
    nkb = (nkb < 17) ? nkb : 17;
    const int mylim = qbase + wave * 16 + 19;
    u16* psw = &ps[wave][0];

#if HAVE_TDM
    if (wave == 0) {
        tdm_load_2d(lds_off(&kbuf[0][0]), kp, 64, 32, 64, KEXT_R, 64);
        __builtin_amdgcn_s_wait_tensorcnt(0);
    }
    __syncthreads();
#endif

    for (int kb = 0; kb < nkb; ++kb) {
        u16* kcur = &kbuf[kb & 1][0];
#if HAVE_TDM
        if (wave == 0 && kb + 1 < nkb)
            tdm_load_2d(lds_off(&kbuf[(kb + 1) & 1][0]),
                        kp + (size_t)(kb + 1) * 32 * 64, 64, 32, 64, KEXT_R, 64);
#else
        __syncthreads();
        {   // cooperative row-major copy of the K block
            const int r = threadIdx.x >> 2, g = threadIdx.x & 3;
            const u16* src = kp + (size_t)(kb * 32 + r) * 64 + g * 16;
            uint4* dst = reinterpret_cast<uint4*>(kcur + r * 64 + g * 16);
            dst[0] = reinterpret_cast<const uint4*>(src)[0];
            dst[1] = reinterpret_cast<const uint4*>(src)[1];
        }
        __syncthreads();
#endif
        if (kb * 32 <= mylim) {
            // scores: two 16-key tiles; K^T B-operands from LDS K tile
            v8f s0 = {}, s1 = {};
            {
                const u16* kr0 = kcur + (0 * 16 + lm) * 64;
                s0 = wmma_bf16(qf0, load16c(kr0 + lh * 16), s0);
                s0 = wmma_bf16(qf1, load16c(kr0 + 32 + lh * 16), s0);
                const u16* kr1 = kcur + (1 * 16 + lm) * 64;
                s1 = wmma_bf16(qf0, load16c(kr1 + lh * 16), s1);
                s1 = wmma_bf16(qf1, load16c(kr1 + 32 + lh * 16), s1);
            }
            const int r0 = kb * 32 + lm, r1 = r0 + 16;
            const int ibase = qbase + wave * 16 + 8 * lh;
            float pv0[8], pv1[8], alpha[8];
#pragma unroll
            for (int p = 0; p < 8; ++p) {
                const int i = ibase + p;
                float a = s0[p] * 0.125f, b = s1[p] * 0.125f;
                if (r0 > i + 4 || r0 >= 516) a = -1e30f;
                if (r1 > i + 4 || r1 >= 516) b = -1e30f;
                float cm = fmaxf(a, b);
                cm = fmaxf(cm, __shfl_xor(cm, 1, 32));
                cm = fmaxf(cm, __shfl_xor(cm, 2, 32));
                cm = fmaxf(cm, __shfl_xor(cm, 4, 32));
                cm = fmaxf(cm, __shfl_xor(cm, 8, 32));
                float mn = fmaxf(mr[p], cm);
                float al = __expf(mr[p] - mn);
                float e0 = __expf(a - mn), e1 = __expf(b - mn);
                float rs = e0 + e1;
                rs += __shfl_xor(rs, 1, 32);
                rs += __shfl_xor(rs, 2, 32);
                rs += __shfl_xor(rs, 4, 32);
                rs += __shfl_xor(rs, 8, 32);
                lr[p] = lr[p] * al + rs;
                mr[p] = mn;
                alpha[p] = al;
                pv0[p] = e0; pv1[p] = e1;
            }
#pragma unroll
            for (int p = 0; p < 8; ++p) {
                o0[p] *= alpha[p]; o1[p] *= alpha[p];
                o2[p] *= alpha[p]; o3[p] *= alpha[p];
            }
            // C-layout P -> row-major 16x32 in per-wave LDS scratch
#pragma unroll
            for (int p = 0; p < 8; ++p) {
                const int row = p + 8 * lh;
                psw[row * 32 + lm]      = f2bf(pv0[p]);
                psw[row * 32 + 16 + lm] = f2bf(pv1[p]);
            }
            __builtin_amdgcn_wave_barrier();
            v16bf pa = load_ab(psw + lm * 32 + lh * 16, psw + lm * 32 + lh * 16 + 8);
            // V^T B-operands straight from global (contiguous 32B rows)
            const u16* vr = vtp + (size_t)lm * KEXT_R + kb * 32 + lh * 16;
            o0 = wmma_bf16(pa, load16c(vr + (size_t)0  * KEXT_R), o0);
            o1 = wmma_bf16(pa, load16c(vr + (size_t)16 * KEXT_R), o1);
            o2 = wmma_bf16(pa, load16c(vr + (size_t)32 * KEXT_R), o2);
            o3 = wmma_bf16(pa, load16c(vr + (size_t)48 * KEXT_R), o3);
        }
#if HAVE_TDM
        if (wave == 0) __builtin_amdgcn_s_wait_tensorcnt(0);
        __syncthreads();
#endif
    }

    // epilogue: O/l -> bf16 attn_out (row = seg*512+i, col = head*64+dh)
    const int ibase = qbase + wave * 16 + 8 * lh;
#pragma unroll
    for (int p = 0; p < 8; ++p) {
        const float inv = 1.0f / lr[p];
        const size_t row = (size_t)seg * 512 + ibase + p;
        u16* dst = aout + row * 512 + head * 64 + lm;
        dst[0]  = f2bf(o0[p] * inv);
        dst[16] = f2bf(o1[p] * inv);
        dst[32] = f2bf(o2[p] * inv);
        dst[48] = f2bf(o3[p] * inv);
    }
}

// ---------- out GEMM: attn_out(16384x512) @ WoutT -> fp32 d_out ----------

__global__ __launch_bounds__(256) void out_gemm_kernel(const u16* __restrict__ a,
                                                       const u16* __restrict__ wt,
                                                       float* __restrict__ out) {
    const int wave = threadIdx.x >> 5, lane = threadIdx.x & 31;
    const int lm = lane & 15, lh = lane >> 4;
    const int m0 = blockIdx.x * 256 + wave * 32;
    const int n0 = blockIdx.y * 64;
    const u16* arow0 = a + (size_t)(m0 + lm) * 512;
    const u16* arow1 = a + (size_t)(m0 + 16 + lm) * 512;
    v8f acc[2][4] = {};
    for (int ks = 0; ks < 16; ++ks) {
        const u16* ap0 = arow0 + ks * 32;
        const u16* ap1 = arow1 + ks * 32;
        if (ks + 1 < 16) { __builtin_prefetch(ap0 + 32, 0, 3); __builtin_prefetch(ap1 + 32, 0, 3); }
        v16bf af0 = load_ab(ap0 + lh * 16, ap0 + lh * 16 + 8);
        v16bf af1 = load_ab(ap1 + lh * 16, ap1 + lh * 16 + 8);
#pragma unroll
        for (int nb = 0; nb < 4; ++nb) {
            const u16* bp = wt + (size_t)(n0 + nb * 16 + lm) * 512 + ks * 32 + lh * 16;
            v16bf bf = load16c(bp);
            acc[0][nb] = wmma_bf16(af0, bf, acc[0][nb]);
            acc[1][nb] = wmma_bf16(af1, bf, acc[1][nb]);
        }
    }
#pragma unroll
    for (int mi = 0; mi < 2; ++mi) {
        const int mb = m0 + mi * 16 + 8 * lh;
#pragma unroll
        for (int nb = 0; nb < 4; ++nb) {
            const int c = n0 + nb * 16 + lm;
#pragma unroll
            for (int p = 0; p < 8; ++p)
                out[(size_t)(mb + p) * 512 + c] = acc[mi][nb][p];
        }
    }
}

// ---------- launcher ----------

extern "C" void kernel_launch(void* const* d_in, const int* in_sizes, int n_in,
                              void* d_out, int out_size, void* d_ws, size_t ws_size,
                              hipStream_t stream) {
    (void)in_sizes; (void)n_in; (void)out_size; (void)ws_size;
    const float* seq  = (const float*)d_in[0];
    const float* rmsw = (const float*)d_in[1];
    const float* wqkv = (const float*)d_in[2];
    const float* wout = (const float*)d_in[3];
    const float* pmem = (const float*)d_in[4];
    float* out  = (float*)d_out;
    float* vout = out + VOUT_OFF;

    char* ws = (char*)d_ws;
    u16* xb    = (u16*)(ws + 0);          // 16384*512*2   = 16,777,216
    u16* wqt   = (u16*)(ws + 16777216);   // 1536*512*2    =  1,572,864
    u16* wot   = (u16*)(ws + 18350080);   // 512*512*2     =    524,288
    u16* qarr  = (u16*)(ws + 18874368);   // 32*8*512*64*2 = 16,777,216
    u16* kext  = (u16*)(ws + 35651584);   // 32*8*544*64*2 = 17,825,792
    u16* vextT = (u16*)(ws + 53477376);   // 32*8*64*544*2 = 17,825,792
    u16* aout  = (u16*)(ws + 71303168);   // 16384*512*2   = 16,777,216

    prep_x_kernel<<<2048, 256, 0, stream>>>(seq, rmsw, xb);
    prep_wT_kernel<<<(512 * 1536 + 255) / 256, 256, 0, stream>>>(wqkv, wqt, 512, 1536);
    prep_wT_kernel<<<(512 * 512 + 255) / 256, 256, 0, stream>>>(wout, wot, 512, 512);
    prep_pmem_kernel<<<(32 * 8 * 32 * 64 + 255) / 256, 256, 0, stream>>>(pmem, kext, vextT);

    qkv_gemm_kernel<<<dim3(64, 24), 256, 0, stream>>>(xb, wqt, qarr, kext, vextT, vout);
    flash_kernel<<<dim3(256, 8), 128, 0, stream>>>(qarr, kext, vextT, aout);
    out_gemm_kernel<<<dim3(64, 8), 256, 0, stream>>>(aout, wot, out);
}
